// GNNEncoder_22084721836817
// MI455X (gfx1250) — matrix operand order repaired and checked by reading, not compiled, not run on hardware
//
#include <hip/hip_runtime.h>
#include <hip/hip_bf16.h>

// ---------------------------------------------------------------------------
// Problem constants (match reference)
// ---------------------------------------------------------------------------
#define NN 100000      // nodes
#define EE 1600000     // edges (before self loops)
#define ET (EE + NN)   // edges incl. self loops
#define GG 64          // graphs
#define HH 4           // heads
#define CC 64          // channels
#define LL 3           // layers
#define ZZ 32          // latent
#define BN_EPS 1e-5f
#define NEG_SLOPE 0.2f
#define MTILES (NN / 16)       // 6250
#define PROJ_BLOCKS 1250       // grid-stride over M tiles

typedef __attribute__((ext_vector_type(16))) _Float16 v16h;
typedef __attribute__((ext_vector_type(8)))  float    v8f;
typedef __attribute__((ext_vector_type(4)))  unsigned u32x4;
typedef __attribute__((ext_vector_type(8)))  int      i32x8;
typedef __attribute__((ext_vector_type(4)))  int      i32x4;

// Monotone float <-> uint32 encoding for atomic max on floats.
__device__ __forceinline__ unsigned f2ord(float f) {
    unsigned u = __float_as_uint(f);
    return (u & 0x80000000u) ? ~u : (u | 0x80000000u);
}
__device__ __forceinline__ float ord2f(unsigned u) {
    return (u & 0x80000000u) ? __uint_as_float(u & 0x7FFFFFFFu)
                             : __uint_as_float(~u);
}
#define ORD_NEG_INF 0x007FFFFFu   // f2ord(-inf)

// ---------------------------------------------------------------------------
// Fill kernels (deterministic per-call init; graph-capture safe)
// ---------------------------------------------------------------------------
__global__ void fill_f32(float* __restrict__ p, float v, int n) {
    int t = blockIdx.x * blockDim.x + threadIdx.x;
    if (t < n) p[t] = v;
}
__global__ void fill_u32(unsigned* __restrict__ p, unsigned v, int n) {
    int t = blockIdx.x * blockDim.x + threadIdx.x;
    if (t < n) p[t] = v;
}

// ---------------------------------------------------------------------------
// A fragment: 16x32 f16 WMMA layout from one row of x (f32), K chunk kb.
// Lane holds 16 elements: [0..15] -> K=kb+klane..+15 (klane = 0 or 16)
// ---------------------------------------------------------------------------
__device__ __forceinline__ v16h load_a_frag(const float4* __restrict__ xr4,
                                            int kb, int klane) {
    float4 p0 = xr4[(kb + klane) >> 2];
    float4 p1 = xr4[((kb + klane) >> 2) + 1];
    float4 p2 = xr4[(kb + 8 + klane) >> 2];
    float4 p3 = xr4[((kb + 8 + klane) >> 2) + 1];
    v16h a;
    a[0] = (_Float16)p0.x;  a[1] = (_Float16)p0.y;
    a[2] = (_Float16)p0.z;  a[3] = (_Float16)p0.w;
    a[4] = (_Float16)p1.x;  a[5] = (_Float16)p1.y;
    a[6] = (_Float16)p1.z;  a[7] = (_Float16)p1.w;
    a[8]  = (_Float16)p2.x; a[9]  = (_Float16)p2.y;
    a[10] = (_Float16)p2.z; a[11] = (_Float16)p2.w;
    a[12] = (_Float16)p3.x; a[13] = (_Float16)p3.y;
    a[14] = (_Float16)p3.z; a[15] = (_Float16)p3.w;
    return a;
}

// ---------------------------------------------------------------------------
// GAT projection: xp[N,256] = x[N,64] @ W[64,256]   (WMMA f16 -> f32)
//
// Block = 512 threads = 16 waves; wave w owns N-tile w (16 of 256 cols).
// W (64KB f32) is DMA'd into LDS once per block via the Tensor Data Mover,
// converted/transposed to f16 [n][k], then each wave hoists its two B
// fragments into registers and grid-strides over M tiles: per tile only
// 8x b128 A loads + 2x v_wmma + 8 stores.
// ---------------------------------------------------------------------------
__global__ __launch_bounds__(512)
void gat_proj_wmma(const float* __restrict__ x, const float* __restrict__ W,
                   float* __restrict__ xp) {
    __shared__ __align__(16) float    Wf32[64 * 256];   // 64 KB raw W
    __shared__ __align__(32) _Float16 Wt[256 * 64];     // 32 KB, Wt[n*64+k]

    const int tid = threadIdx.x;

#if defined(__has_builtin) && __has_builtin(__builtin_amdgcn_tensor_load_to_lds)
    // ---- Tensor Data Mover: one DMA of the 64x256 f32 tile into LDS ----
    if (tid < 32) {                       // single wave issues (TENSORcnt per-wave)
        size_t   wa      = (size_t)W;
        unsigned lds_off = (unsigned)(size_t)(void*)&Wf32[0];
        u32x4 g0 = { 1u,                                  // count=1, user mode
                     lds_off,                             // lds_addr
                     (unsigned)(wa & 0xFFFFFFFFu),        // global_addr[31:0]
                     (unsigned)((wa >> 32) & 0x01FFFFFFu) // global_addr[56:32]
                       | (2u << 30) };                    // type=2 (image)
        i32x8 g1 = { 0x00020000,   // data_size=2 (4B), no mask/pad/iterate
                     0x01000000,   // tensor_dim0=256 (bits 63:48)
                     0x00400000,   // tensor_dim1=64  (bits 95:80)
                     0x01000000,   // tile_dim0=256   (bits 127:112)
                     0x00000040,   // tile_dim1=64    (bits 143:128)
                     0x00000100,   // tensor_dim0_stride=256 (bits 191:160)
                     0, 0 };
        i32x4 gz4 = { 0, 0, 0, 0 };      // groups 2/3 unused (2-D tensor)
        i32x8 gz8 = { 0, 0, 0, 0, 0, 0, 0, 0 };
        __builtin_amdgcn_tensor_load_to_lds(g0, g1, gz4, gz4, gz8, 0);
        __builtin_amdgcn_s_wait_tensorcnt(0);
    }
    asm volatile("" ::: "memory");        // LDS now holds W; don't assume undef
    __syncthreads();
#else
    for (int i = tid; i < 64 * 256; i += 512) Wf32[i] = W[i];
    __syncthreads();
#endif

    // convert + transpose to f16: Wt[n][k]
    for (int i = tid; i < 256 * 64; i += 512) {
        int k = i >> 8;       // 0..63
        int n = i & 255;      // 0..255
        Wt[n * 64 + k] = (_Float16)Wf32[k * 256 + n];
    }
    __syncthreads();

    const int wave  = tid >> 5;                 // 0..15 -> N tile
    const int lane  = tid & 31;
    const int klane = (lane < 16) ? 0 : 16;     // A K sub-offset
    const int ncol  = wave * 16 + (lane & 15);  // B-matrix column
    const int klo   = (lane < 16) ? 0 : 16;     // B K sub-offset
    const int nOut  = ncol;
    const int mOff  = (lane < 16) ? 0 : 8;

    // Hoist B fragments (invariant across M tiles): 32B contiguous LDS reads
    v16h b0 = *(const v16h*)&Wt[ncol * 64 + 0  + klo];
    v16h b1 = *(const v16h*)&Wt[ncol * 64 + 32 + klo];

    for (int mt = blockIdx.x; mt < MTILES; mt += gridDim.x) {
        const int row = mt * 16 + (lane & 15);
        const float4* xr4 = (const float4*)(x + (size_t)row * CC);
        // prefetch next M tile's rows (speculative; dropped if OOB)
        __builtin_prefetch(x + (size_t)(row + (int)gridDim.x * 16) * CC, 0, 0);

        v16h a0 = load_a_frag(xr4, 0, klane);
        v16h a1 = load_a_frag(xr4, 32, klane);
        v8f  c  = {};
        c = __builtin_amdgcn_wmma_f32_16x16x32_f16(false, a0, false, b0,
                                                   (short)0, c, false, false);
        c = __builtin_amdgcn_wmma_f32_16x16x32_f16(false, a1, false, b1,
                                                   (short)0, c, false, false);

        // D layout: VGPR r -> M = r + mOff, N = lane&15
        const int mBase = mt * 16 + mOff;
        #pragma unroll
        for (int r = 0; r < 8; ++r)
            xp[(size_t)(mBase + r) * 256 + nOut] = c[r];
    }
}

// ---------------------------------------------------------------------------
// alpha_src / alpha_dst: [N,H] = einsum('nhc,hc->nh')
// ---------------------------------------------------------------------------
__global__ void gat_alpha(const float* __restrict__ xp,
                          const float* __restrict__ a_src,
                          const float* __restrict__ a_dst,
                          float* __restrict__ als, float* __restrict__ ald) {
    int t = blockIdx.x * blockDim.x + threadIdx.x;   // n*H + h
    if (t >= NN * HH) return;
    int n = t >> 2, h = t & 3;
    const float* v  = xp + (size_t)n * 256 + h * CC;
    const float* as = a_src + h * CC;
    const float* ad = a_dst + h * CC;
    float ss = 0.f, sd = 0.f;
    #pragma unroll 8
    for (int cc = 0; cc < CC; ++cc) { float xv = v[cc]; ss += xv * as[cc]; sd += xv * ad[cc]; }
    als[t] = ss; ald[t] = sd;
}

__device__ __forceinline__ void edge_ends(int e, const int* __restrict__ src,
                                          const int* __restrict__ dst,
                                          int& s, int& d) {
    if (e < EE) { s = src[e]; d = dst[e]; }
    else        { s = e - EE; d = e - EE; }          // self-loop
}

// segment max of leaky-relu edge logits
__global__ void edge_max(const int* __restrict__ src, const int* __restrict__ dst,
                         const float* __restrict__ als, const float* __restrict__ ald,
                         unsigned* __restrict__ mbits) {
    int t = blockIdx.x * blockDim.x + threadIdx.x;   // e*H + h
    if (t >= ET * HH) return;
    int e = t >> 2, h = t & 3, s, d;
    edge_ends(e, src, dst, s, d);
    float v = als[s * HH + h] + ald[d * HH + h];
    v = (v > 0.f) ? v : NEG_SLOPE * v;
    atomicMax(&mbits[d * HH + h], f2ord(v));
}

// exp(e - max) and segment sum
__global__ void edge_expsum(const int* __restrict__ src, const int* __restrict__ dst,
                            const float* __restrict__ als, const float* __restrict__ ald,
                            const unsigned* __restrict__ mbits,
                            float* __restrict__ exbuf, float* __restrict__ ssum) {
    int t = blockIdx.x * blockDim.x + threadIdx.x;
    if (t >= ET * HH) return;
    int e = t >> 2, h = t & 3, s, d;
    edge_ends(e, src, dst, s, d);
    float v = als[s * HH + h] + ald[d * HH + h];
    v = (v > 0.f) ? v : NEG_SLOPE * v;
    float ex = __expf(v - ord2f(mbits[d * HH + h]));
    exbuf[t] = ex;
    atomicAdd(&ssum[d * HH + h], ex);
}

// weighted scatter: acc[dst,h,c] += xp[src,h,c] * a
// 64 threads per edge, 4 channels (one b128 gather) per thread.
__global__ __launch_bounds__(256)
void edge_scatter(const int* __restrict__ src, const int* __restrict__ dst,
                  const float* __restrict__ exbuf, const float* __restrict__ ssum,
                  const float* __restrict__ xp, float* __restrict__ acc) {
    int e = blockIdx.x * 4 + (threadIdx.x >> 6);    // edge index
    if (e >= ET) return;
    int q  = threadIdx.x & 63;                      // 0..63
    int hc = q * 4;                                 // channel base (h*64+c)
    int h  = q >> 4;
    int s, d;
    edge_ends(e, src, dst, s, d);
    float a = exbuf[e * HH + h] / ssum[d * HH + h];
    const float4 xv = *(const float4*)&xp[(size_t)s * 256 + hc];
    float* ap = &acc[(size_t)d * 256 + hc];
    atomicAdd(ap + 0, xv.x * a);
    atomicAdd(ap + 1, xv.y * a);
    atomicAdd(ap + 2, xv.z * a);
    atomicAdd(ap + 3, xv.w * a);
}

// head mean + bias + BN(eval) + ReLU
__global__ void layer_finalize(const float* __restrict__ acc,
                               const float* __restrict__ bias,
                               const float* __restrict__ gamma,
                               const float* __restrict__ beta,
                               const float* __restrict__ mean,
                               const float* __restrict__ var,
                               float* __restrict__ xout) {
    int t = blockIdx.x * blockDim.x + threadIdx.x;   // n*C + c
    if (t >= NN * CC) return;
    int c = t & 63;
    const float* p = acc + ((size_t)(t >> 6)) * 256 + c;
    float v = 0.25f * (p[0] + p[64] + p[128] + p[192]) + bias[c];
    v = (v - mean[c]) * (gamma[c] * __frsqrt_rn(var[c] + BN_EPS)) + beta[c];
    xout[t] = (v > 0.f) ? v : 0.f;
}

// graph pooling: mean-sum + max per (graph, channel)
__global__ void pool(const float* __restrict__ x, const int* __restrict__ batch,
                     float* __restrict__ gsum, unsigned* __restrict__ gmaxb,
                     float* __restrict__ cnt) {
    int t = blockIdx.x * blockDim.x + threadIdx.x;   // n*C + c
    if (t >= NN * CC) return;
    int n = t >> 6, c = t & 63;
    int g = batch[n];
    float v = x[t];
    atomicAdd(&gsum[g * CC + c], v);
    atomicMax(&gmaxb[g * CC + c], f2ord(v));
    if (c == 0) atomicAdd(&cnt[g], 1.0f);
}

// VAE head: h=[mean||max] @ {mu,logvar} weights  (tiny: 2048 threads x 128 MAC)
__global__ void vae_head(const float* __restrict__ gsum, const unsigned* __restrict__ gmaxb,
                         const float* __restrict__ cnt,
                         const float* __restrict__ muW, const float* __restrict__ muB,
                         const float* __restrict__ lvW, const float* __restrict__ lvB,
                         float* __restrict__ out) {
    int t = blockIdx.x * blockDim.x + threadIdx.x;   // g*Z + z
    if (t >= GG * ZZ) return;
    int g = t >> 5, z = t & 31;
    float c_ = cnt[g];
    float invc = 1.0f / (c_ > 1.0f ? c_ : 1.0f);
    float smu = muB[z], slv = lvB[z];
    #pragma unroll 4
    for (int j = 0; j < 2 * CC; ++j) {
        float hv = (j < CC) ? gsum[g * CC + j] * invc
                            : ord2f(gmaxb[g * CC + (j - CC)]);
        smu += hv * muW[j * ZZ + z];
        slv += hv * lvW[j * ZZ + z];
    }
    out[t]           = smu;   // mu     [G,Z]
    out[GG * ZZ + t] = slv;   // logvar [G,Z]
}

// ---------------------------------------------------------------------------
// Host orchestration
// ---------------------------------------------------------------------------
static inline dim3 g1(long long n, int b) { return dim3((unsigned)((n + b - 1) / b)); }

extern "C" void kernel_launch(void* const* d_in, const int* in_sizes, int n_in,
                              void* d_out, int out_size, void* d_ws, size_t ws_size,
                              hipStream_t stream) {
    const float* x0     = (const float*)d_in[0];
    const int*   eidx   = (const int*)d_in[1];
    const int*   batch  = (const int*)d_in[2];
    const float* W      = (const float*)d_in[3];   // [L,64,256]
    const float* attS   = (const float*)d_in[4];   // [L,H,C]
    const float* attD   = (const float*)d_in[5];
    const float* gbias  = (const float*)d_in[6];   // [L,C]
    const float* bng    = (const float*)d_in[7];
    const float* bnb    = (const float*)d_in[8];
    const float* bnm    = (const float*)d_in[9];
    const float* bnv    = (const float*)d_in[10];
    const float* muW    = (const float*)d_in[11];
    const float* muB    = (const float*)d_in[12];
    const float* lvW    = (const float*)d_in[13];
    const float* lvB    = (const float*)d_in[14];
    float* out = (float*)d_out;

    const int* src = eidx;        // edge_index[0]
    const int* dst = eidx + EE;   // edge_index[1]

    // workspace layout
    float*    xp    = (float*)d_ws;                 // [N,256]
    float*    acc   = xp   + (size_t)NN * 256;      // [N,256]
    float*    bufA  = acc  + (size_t)NN * 256;      // [N,64]
    float*    bufB  = bufA + (size_t)NN * CC;       // [N,64]
    float*    als   = bufB + (size_t)NN * CC;       // [N,H]
    float*    ald   = als  + (size_t)NN * HH;
    unsigned* mbits = (unsigned*)(ald + (size_t)NN * HH);
    float*    ssum  = (float*)(mbits + (size_t)NN * HH);
    float*    exbuf = ssum + (size_t)NN * HH;       // [ET,H]
    float*    gsum  = exbuf + (size_t)ET * HH;      // [G,C]
    unsigned* gmaxb = (unsigned*)(gsum + GG * CC);
    float*    cnt   = (float*)(gmaxb + GG * CC);    // [G]

    const float* xin = x0;
    for (int i = 0; i < LL; ++i) {
        float* xout = (i & 1) ? bufB : bufA;

        gat_proj_wmma<<<PROJ_BLOCKS, 512, 0, stream>>>(
            xin, W + (size_t)i * CC * 256, xp);
        gat_alpha<<<g1((long long)NN * HH, 256), 256, 0, stream>>>(
            xp, attS + i * HH * CC, attD + i * HH * CC, als, ald);

        fill_u32<<<g1(NN * HH, 256), 256, 0, stream>>>(mbits, ORD_NEG_INF, NN * HH);
        fill_f32<<<g1(NN * HH, 256), 256, 0, stream>>>(ssum, 0.f, NN * HH);
        fill_f32<<<g1((long long)NN * 256, 256), 256, 0, stream>>>(acc, 0.f, NN * 256);

        edge_max<<<g1((long long)ET * HH, 256), 256, 0, stream>>>(src, dst, als, ald, mbits);
        edge_expsum<<<g1((long long)ET * HH, 256), 256, 0, stream>>>(
            src, dst, als, ald, mbits, exbuf, ssum);
        edge_scatter<<<g1(ET, 4), 256, 0, stream>>>(src, dst, exbuf, ssum, xp, acc);

        layer_finalize<<<g1((long long)NN * CC, 256), 256, 0, stream>>>(
            acc, gbias + i * CC, bng + i * CC, bnb + i * CC,
            bnm + i * CC, bnv + i * CC, xout);
        xin = xout;
    }

    fill_f32<<<g1(GG * CC, 256), 256, 0, stream>>>(gsum, 0.f, GG * CC);
    fill_u32<<<g1(GG * CC, 256), 256, 0, stream>>>(gmaxb, ORD_NEG_INF, GG * CC);
    fill_f32<<<1, GG, 0, stream>>>(cnt, 0.f, GG);

    pool<<<g1((long long)NN * CC, 256), 256, 0, stream>>>(xin, batch, gsum, gmaxb, cnt);
    vae_head<<<g1(GG * ZZ, 256), 256, 0, stream>>>(gsum, gmaxb, cnt, muW, muB, lvW, lvB, out);
}
